// SparseTechnicalNetwork_28441273434822
// MI455X (gfx1250) — matrix operand order repaired and checked by reading, not compiled, not run on hardware
//
#include <hip/hip_runtime.h>
#include <cstdint>

// ---------------------------------------------------------------------------
// Problem constants
// ---------------------------------------------------------------------------
#define NB      512          // batch
#define TT      60           // timesteps
#define FF      5            // input features
#define HH      128          // hidden
#define GG      512          // 4*H gates
#define NN      2500         // neurons
#define NPAD    2528         // padded to multiple of 32 for WMMA K
#define BT      (NB*TT)      // 30720

typedef __attribute__((ext_vector_type(16))) __bf16 v16bf;
typedef __attribute__((ext_vector_type(8)))  float  v8f;

union FragBF { v16bf v; uint4 q[2]; };

__device__ __forceinline__ unsigned short f2bf(float x) {
    unsigned int u = __float_as_uint(x);
    unsigned int r = u + 0x7FFFu + ((u >> 16) & 1u);   // round-to-nearest-even
    return (unsigned short)(r >> 16);
}
__device__ __forceinline__ float bf2f(unsigned short h) {
    return __uint_as_float(((unsigned int)h) << 16);
}
__device__ __forceinline__ float sigf(float x) { return 1.0f / (1.0f + __expf(-x)); }

// ---------------------------------------------------------------------------
// Kernel 1: gates0[b,t,:] = x[b,t,:] @ W_ih0.T + b_ih0 + b_hh0   (K=5, VALU)
// ---------------------------------------------------------------------------
__global__ void gates0_kernel(const float* __restrict__ x,
                              const float* __restrict__ Wih0,
                              const float* __restrict__ bih,
                              const float* __restrict__ bhh,
                              float* __restrict__ g0) {
    size_t idx = (size_t)blockIdx.x * 256 + threadIdx.x;
    if (idx >= (size_t)BT * GG) return;
    int n = (int)(idx & (GG - 1));
    size_t bt = idx >> 9;
    const float* xp = x + bt * FF;
    const float* wp = Wih0 + (size_t)n * FF;
    float s = bih[n] + bhh[n];
#pragma unroll
    for (int f = 0; f < FF; ++f) s += xp[f] * wp[f];
    g0[idx] = s;
}

// ---------------------------------------------------------------------------
// Kernel 2: misc prep (all flat):
//   [0, 256*NPAD)         : int_w1 -> bf16, zero-padded K
//   next 512*128          : W_ih1 -> bf16
//   next 2500             : cs[n] = sum_k conn_w[n,k]
//   next 512              : bias1[n] = b_ih1[n] + b_hh1[n]
// ---------------------------------------------------------------------------
__global__ void prep_misc_kernel(const float* __restrict__ int_w1,
                                 const float* __restrict__ Wih1,
                                 const float* __restrict__ conn_w,
                                 const float* __restrict__ bih1,
                                 const float* __restrict__ bhh1,
                                 unsigned short* __restrict__ intw1bf,
                                 unsigned short* __restrict__ wih1bf,
                                 float* __restrict__ cs,
                                 float* __restrict__ bias1) {
    const size_t R0 = (size_t)256 * NPAD;          // 647168
    const size_t R1 = R0 + (size_t)GG * HH;        // +65536
    const size_t R2 = R1 + NN;                     // +2500
    const size_t R3 = R2 + GG;                     // +512
    size_t idx = (size_t)blockIdx.x * 256 + threadIdx.x;
    if (idx < R0) {
        int row = (int)(idx / NPAD), col = (int)(idx % NPAD);
        float v = (col < NN) ? int_w1[(size_t)row * NN + col] : 0.0f;
        intw1bf[idx] = f2bf(v);
    } else if (idx < R1) {
        size_t j = idx - R0;
        wih1bf[j] = f2bf(Wih1[j]);
    } else if (idx < R2) {
        int n = (int)(idx - R1);
        float s = 0.0f;
        for (int k = 0; k < 50; ++k) s += conn_w[(size_t)n * 50 + k];
        cs[n] = s;
    } else if (idx < R3) {
        int n = (int)(idx - R2);
        bias1[n] = bih1[n] + bhh1[n];
    }
}

// ---------------------------------------------------------------------------
// Kernel 3: LSTM recurrence with LDS-resident Whh (bf16) and WMMA.
//   One workgroup = 64 batch rows, 256 threads = 8 waves.
//   Whh staged to LDS reordered as [nb][gate][16 rows][128 k] so that each
//   wave-group's 4 gate B-fragments fit in the 16-bit DS immediate window
//   from a single base register (no address spills).
//   h is double-buffered in LDS (bf16): one barrier per timestep, the
//   elementwise phase writes h(t+1) straight into the other buffer.
//   Cell state c lives in registers (layout == WMMA C-fragment layout).
// ---------------------------------------------------------------------------
__global__ void lstm_rec_kernel(const float* __restrict__ gates_in,
                                const float* __restrict__ Whh,
                                unsigned short* __restrict__ y_out,
                                float* __restrict__ hlast_out) {
    extern __shared__ unsigned char smem_raw[];
    unsigned short* whh_s = (unsigned short*)smem_raw;        // [8][4][16][128] bf16 (128 KB)
    unsigned short* h0_s  = whh_s + GG * HH;                  // [64][128] bf16 (16 KB)
    unsigned short* h1_s  = h0_s + 64 * HH;                   // [64][128] bf16 (16 KB)

    const int tid  = threadIdx.x;
    const int lane = tid & 31;
    const int wave = tid >> 5;
    const int l15  = lane & 15;
    const int half = lane >> 4;       // 0/1
    const int crow = half << 3;       // C-fragment M offset (0 or 8)
    const int akof = half << 3;       // A-fragment K chunk offset (0 or 8)
    const int bkof = half << 3;       // B-fragment K chunk offset (0 or 8)
    const int bbase = blockIdx.x * 64;

    // stage Whh -> LDS bf16, reordered [nb][gate][row16][k]
    for (int i = tid; i < GG * HH; i += 256) {
        int n    = i >> 7;        // 0..511 (original gate-major row)
        int k    = i & 127;
        int gate = n >> 7;        // 0..3
        int rem  = n & 127;
        int nb   = rem >> 4;      // 0..7
        int r16  = rem & 15;
        int ni = ((nb * 4 + gate) * 16 + r16) * HH + k;
        whh_s[ni] = f2bf(Whh[i]);
    }
    for (int i = tid; i < 64 * HH; i += 256) h0_s[i] = 0;

    // per-group (this wave owns 4 groups) constants
    int mi_[4], nb_[4], aoff[4], woff[4];
    const unsigned short* bB[4];
#pragma unroll
    for (int q = 0; q < 4; ++q) {
        int gidx = wave * 4 + q;
        mi_[q] = gidx >> 3;           // 0..3 (16-row tile)
        nb_[q] = gidx & 7;            // 0..7 (16-col group within H)
        bB[q]  = whh_s + (size_t)nb_[q] * 8192 + l15 * HH + bkof;  // per-nb slab
        aoff[q] = (mi_[q] * 16 + l15) * HH + akof;
        woff[q] = (mi_[q] * 16 + crow) * HH + nb_[q] * 16 + l15;
    }

    float cst[4][8];
#pragma unroll
    for (int q = 0; q < 4; ++q)
#pragma unroll
        for (int r = 0; r < 8; ++r) cst[q][r] = 0.0f;

    for (int t = 0; t < TT; ++t) {
        __syncthreads();                       // h(t) buffer complete
        unsigned short* cur = (t & 1) ? h1_s : h0_s;
        unsigned short* nxt = (t & 1) ? h0_s : h1_s;

#pragma unroll
        for (int q = 0; q < 4; ++q) {
            const int rowg0 = bbase + mi_[q] * 16;

            // init accumulators from the precomputed x-gates (acts as C input);
            // all 32 loads share one base VGPR pair, offsets in the 24-bit imm
            v8f acc[4];
            const float* gp = gates_in
                + ((size_t)(rowg0 + crow) * TT + t) * GG + nb_[q] * 16 + l15;
#pragma unroll
            for (int gate = 0; gate < 4; ++gate)
#pragma unroll
                for (int r = 0; r < 8; ++r)
                    acc[gate][r] = gp[(size_t)r * TT * GG + gate * HH];

            // K loop: 128 = 4 x 32 ; A fragment shared across the 4 gates;
            // B fragments all at immediate offsets from bB[q]
            const unsigned short* ap0 = cur + aoff[q];
            const unsigned short* bp0 = bB[q];
#pragma unroll
            for (int kk = 0; kk < 4; ++kk) {
                const int kb = kk * 32;
                FragBF a;
                a.q[0] = *(const uint4*)(ap0 + kb);
                a.q[1] = *(const uint4*)(ap0 + kb + 16);
#pragma unroll
                for (int gate = 0; gate < 4; ++gate) {
                    FragBF b;
                    const unsigned short* bp = bp0 + gate * 2048 + kb;
                    b.q[0] = *(const uint4*)(bp);
                    b.q[1] = *(const uint4*)(bp + 16);
                    acc[gate] = __builtin_amdgcn_wmma_f32_16x16x32_bf16(
                        false, a.v, false, b.v, (short)0, acc[gate], false, false);
                }
            }

            // element-wise LSTM cell: gate order [i, f, g, o];
            // write h(t+1) straight into the other LDS buffer
            unsigned short* wp = nxt + woff[q];
#pragma unroll
            for (int r = 0; r < 8; ++r) {
                float iv = sigf(acc[0][r]);
                float fv = sigf(acc[1][r]);
                float gv = tanhf(acc[2][r]);
                float ov = sigf(acc[3][r]);
                float c  = fv * cst[q][r] + iv * gv;
                cst[q][r] = c;
                float hv = ov * tanhf(c);
                unsigned short hb = f2bf(hv);
                wp[r * HH] = hb;                         // ds_store_b16, imm r*256
                const int rowg = rowg0 + r + crow;
                const int col  = nb_[q] * 16 + l15;
                if (y_out)
                    y_out[((size_t)rowg * TT + t) * HH + col] = hb;
                if (hlast_out && t == TT - 1)
                    hlast_out[(size_t)rowg * HH + col] = hv;
            }
        }
    }
}

// ---------------------------------------------------------------------------
// Kernel 4: generic bf16 WMMA GEMM: C[M][N] = act(A[M][K] @ W[N][K].T + bias)
//   128 threads = 4 waves; one wave computes a 64x16 tile strip (4 M-tiles
//   sharing each B fragment -> 4x less B traffic). M%64==0, N%16==0, K%32==0.
// ---------------------------------------------------------------------------
__global__ void wmma_gemm_bf16(const unsigned short* __restrict__ A,
                               const unsigned short* __restrict__ W,
                               const float* __restrict__ bias,
                               float* __restrict__ C,
                               int M, int N, int K, int act) {
    const int tid  = threadIdx.x;
    const int lane = tid & 31;
    const int wave = tid >> 5;
    const int l15  = lane & 15;
    const int half = lane >> 4;
    const int crow = half << 3;
    const int akof = half << 3;
    const int bkof = half << 3;

    const int tiles_n = N >> 4;
    const int mblocks = M >> 6;                 // 64-row strips
    const int grp = blockIdx.x * 4 + wave;
    if (grp >= mblocks * tiles_n) return;
    const int gm = grp / tiles_n;
    const int tn = grp % tiles_n;

    v8f acc[4];
#pragma unroll
    for (int s = 0; s < 4; ++s)
#pragma unroll
        for (int r = 0; r < 8; ++r) acc[s][r] = 0.0f;

    const unsigned short* Ab[4];
#pragma unroll
    for (int s = 0; s < 4; ++s)
        Ab[s] = A + (size_t)(gm * 64 + s * 16 + l15) * K + akof;
    const unsigned short* Wb = W + (size_t)(tn * 16 + l15) * K + bkof;

    for (int kb = 0; kb < K; kb += 32) {
        FragBF b;
        b.q[0] = *(const uint4*)(Wb + kb);
        b.q[1] = *(const uint4*)(Wb + kb + 16);
#pragma unroll
        for (int s = 0; s < 4; ++s) {
            FragBF a;
            a.q[0] = *(const uint4*)(Ab[s] + kb);
            a.q[1] = *(const uint4*)(Ab[s] + kb + 16);
            acc[s] = __builtin_amdgcn_wmma_f32_16x16x32_bf16(
                false, a.v, false, b.v, (short)0, acc[s], false, false);
        }
    }

    const int col = tn * 16 + l15;
    const float bv = bias ? bias[col] : 0.0f;
#pragma unroll
    for (int s = 0; s < 4; ++s) {
        const int row0 = gm * 64 + s * 16 + crow;
#pragma unroll
        for (int r = 0; r < 8; ++r) {
            float v = acc[s][r] + bv;
            if (act == 1) v = fmaxf(v, 0.0f);
            C[(size_t)(row0 + r) * N + col] = v;
        }
    }
}

// ---------------------------------------------------------------------------
// Kernel 5: feature pipe: feat[b] = sum(tanh(relu(h@fp_w1.T+b1)@fp_w2.T+b2))
//   one block (64 threads) per batch row
// ---------------------------------------------------------------------------
__global__ void feat_kernel(const float* __restrict__ hlast,
                            const float* __restrict__ fp_w1,
                            const float* __restrict__ fp_b1,
                            const float* __restrict__ fp_w2,
                            const float* __restrict__ fp_b2,
                            float* __restrict__ feat) {
    __shared__ float h1s[64];
    __shared__ float bs[32];
    const int b = blockIdx.x, j = threadIdx.x;
    const float* hr = hlast + (size_t)b * HH;
    float s = fp_b1[j];
    for (int k = 0; k < HH; ++k) s += hr[k] * fp_w1[(size_t)j * HH + k];
    h1s[j] = fmaxf(s, 0.0f);
    __syncthreads();
    if (j < 32) {
        float s2 = fp_b2[j];
        for (int k = 0; k < 64; ++k) s2 += h1s[k] * fp_w2[(size_t)j * 64 + k];
        bs[j] = tanhf(s2);
    }
    __syncthreads();
    if (j == 0) {
        float t = 0.0f;
        for (int k = 0; k < 32; ++k) t += bs[k];
        feat[b] = t;
    }
}

// ---------------------------------------------------------------------------
// Kernel 6: neuron activations -> bf16 padded [B][NPAD]
// ---------------------------------------------------------------------------
__global__ void neuron_kernel(const float* __restrict__ feat,
                              const float* __restrict__ cs,
                              const float* __restrict__ sens,
                              const float* __restrict__ thr,
                              unsigned short* __restrict__ neuron) {
    size_t idx = (size_t)blockIdx.x * 256 + threadIdx.x;
    if (idx >= (size_t)NB * NPAD) return;
    const int b = (int)(idx / NPAD), n = (int)(idx % NPAD);
    float v = 0.0f;
    if (n < NN) {
        float pre = feat[b] * cs[n] * sens[n];
        if (n < 800)       v = sigf(pre - thr[n]);
        else if (n < 1500) v = tanhf(pre);
        else if (n < 2100) v = fmaxf(pre - thr[n], 0.0f);
        else               v = sigf(pre);
    }
    neuron[idx] = f2bf(v);
}

// ---------------------------------------------------------------------------
// Kernel 7: tail per-row: z2 = relu(z1@int_w2.T+b), integ = tanh(z2@int_w3.T+b),
//   5 heads, sigmoid(conf), 4 group means. One block (256 thr) per row.
// ---------------------------------------------------------------------------
__global__ void head_kernel(const float* __restrict__ z1,
                            const float* __restrict__ int_w2,
                            const float* __restrict__ int_b2,
                            const float* __restrict__ int_w3,
                            const float* __restrict__ int_b3,
                            const float* __restrict__ tw, const float* __restrict__ tb,
                            const float* __restrict__ pw, const float* __restrict__ pb,
                            const float* __restrict__ kwp, const float* __restrict__ kbp,
                            const float* __restrict__ vw, const float* __restrict__ vb,
                            const float* __restrict__ cw, const float* __restrict__ cb,
                            const unsigned short* __restrict__ neuron,
                            float* __restrict__ out) {
    __shared__ float z2s[64];
    __shared__ float integs[32];
    __shared__ float gsum[4];
    const int b = blockIdx.x, j = threadIdx.x;
    const float* z1r = z1 + (size_t)b * 256;

    if (j < 64) {
        float s = int_b2[j];
        for (int k = 0; k < 256; ++k) s += z1r[k] * int_w2[(size_t)j * 256 + k];
        z2s[j] = fmaxf(s, 0.0f);
    }
    if (j < 4) gsum[j] = 0.0f;
    __syncthreads();
    if (j < 32) {
        float s = int_b3[j];
        for (int k = 0; k < 64; ++k) s += z2s[k] * int_w3[(size_t)j * 64 + k];
        integs[j] = tanhf(s);
    }
    __syncthreads();

    float* o = out + (size_t)b * 20;
    if (j < 15) {
        const float* w; const float* bb; int r;
        if (j < 3)       { w = tw;  bb = tb;  r = j; }
        else if (j < 9)  { w = pw;  bb = pb;  r = j - 3; }
        else if (j < 13) { w = kwp; bb = kbp; r = j - 9; }
        else if (j == 13){ w = vw;  bb = vb;  r = 0; }
        else             { w = cw;  bb = cb;  r = 0; }
        float s = bb[r];
        for (int k = 0; k < 32; ++k) s += integs[k] * w[r * 32 + k];
        o[j] = s;
        if (j == 14) o[15] = sigf(s);
    }

    // group means over neuron_out
    const unsigned short* nr = neuron + (size_t)b * NPAD;
    float l0 = 0, l1 = 0, l2 = 0, l3 = 0;
    for (int n = j; n < NN; n += 256) {
        float v = bf2f(nr[n]);
        if (n < 800)       l0 += v;
        else if (n < 1500) l1 += v;
        else if (n < 2100) l2 += v;
        else               l3 += v;
    }
    atomicAdd(&gsum[0], l0);
    atomicAdd(&gsum[1], l1);
    atomicAdd(&gsum[2], l2);
    atomicAdd(&gsum[3], l3);
    __syncthreads();
    if (j == 0) {
        o[16] = gsum[0] / 800.0f;
        o[17] = gsum[1] / 700.0f;
        o[18] = gsum[2] / 600.0f;
        o[19] = gsum[3] / 400.0f;
    }
}

// ---------------------------------------------------------------------------
// Host launcher
// ---------------------------------------------------------------------------
extern "C" void kernel_launch(void* const* d_in, const int* in_sizes, int n_in,
                              void* d_out, int out_size, void* d_ws, size_t ws_size,
                              hipStream_t stream) {
    const float* x      = (const float*)d_in[0];
    const float* W_ih0  = (const float*)d_in[1];
    const float* W_hh0  = (const float*)d_in[2];
    const float* b_ih0  = (const float*)d_in[3];
    const float* b_hh0  = (const float*)d_in[4];
    const float* W_ih1  = (const float*)d_in[5];
    const float* W_hh1  = (const float*)d_in[6];
    const float* b_ih1  = (const float*)d_in[7];
    const float* b_hh1  = (const float*)d_in[8];
    const float* fp_w1  = (const float*)d_in[9];
    const float* fp_b1  = (const float*)d_in[10];
    const float* fp_w2  = (const float*)d_in[11];
    const float* fp_b2  = (const float*)d_in[12];
    const float* conn_w = (const float*)d_in[13];
    const float* sens   = (const float*)d_in[14];
    const float* thr    = (const float*)d_in[15];
    const float* int_w1 = (const float*)d_in[16];
    const float* int_b1 = (const float*)d_in[17];
    const float* int_w2 = (const float*)d_in[18];
    const float* int_b2 = (const float*)d_in[19];
    const float* int_w3 = (const float*)d_in[20];
    const float* int_b3 = (const float*)d_in[21];
    const float* tw     = (const float*)d_in[22];
    const float* tb     = (const float*)d_in[23];
    const float* pw     = (const float*)d_in[24];
    const float* pb     = (const float*)d_in[25];
    const float* kwp    = (const float*)d_in[26];
    const float* kbp    = (const float*)d_in[27];
    const float* vw     = (const float*)d_in[28];
    const float* vb     = (const float*)d_in[29];
    const float* cw     = (const float*)d_in[30];
    const float* cb     = (const float*)d_in[31];
    // d_in[32] = conn_idx: provably unused (acts is a broadcast scalar per row)

    // ---- workspace carve-up ----
    char* ws = (char*)d_ws;
    size_t off = 0;
    auto carve = [&](size_t bytes) -> char* {
        char* p = ws + off;
        off = (off + bytes + 255) & ~(size_t)255;
        return p;
    };
    float*          g0buf   = (float*)         carve((size_t)BT * GG * 4);   // gates (reused for g1)
    unsigned short* y0buf   = (unsigned short*)carve((size_t)BT * HH * 2);   // layer0 output bf16
    float*          hlast   = (float*)         carve((size_t)NB * HH * 4);
    unsigned short* wih1bf  = (unsigned short*)carve((size_t)GG * HH * 2);
    float*          bias1   = (float*)         carve((size_t)GG * 4);
    unsigned short* intw1bf = (unsigned short*)carve((size_t)256 * NPAD * 2);
    unsigned short* neuron  = (unsigned short*)carve((size_t)NB * NPAD * 2);
    float*          z1      = (float*)         carve((size_t)NB * 256 * 4);
    float*          cs      = (float*)         carve((size_t)NN * 4);
    float*          feat    = (float*)         carve((size_t)NB * 4);
    (void)ws_size; (void)in_sizes; (void)n_in; (void)out_size;

    float* out = (float*)d_out;
    const size_t lstm_smem = (size_t)(GG * HH + 2 * 64 * HH) * 2;   // 160 KB

    // 1. precompute layer0 x-gates
    {
        size_t total = (size_t)BT * GG;
        gates0_kernel<<<(unsigned)((total + 255) / 256), 256, 0, stream>>>(
            x, W_ih0, b_ih0, b_hh0, g0buf);
    }
    // 2. misc prep (bf16 weight staging, conn_w row sums, combined bias1)
    {
        size_t total = (size_t)256 * NPAD + (size_t)GG * HH + NN + GG;
        prep_misc_kernel<<<(unsigned)((total + 255) / 256), 256, 0, stream>>>(
            int_w1, W_ih1, conn_w, b_ih1, b_hh1, intw1bf, wih1bf, cs, bias1);
    }
    // 3. LSTM layer 0 (WMMA recurrence), writes y0 bf16
    lstm_rec_kernel<<<NB / 64, 256, lstm_smem, stream>>>(g0buf, W_hh0, y0buf, nullptr);
    // 4. g1 = y0 @ W_ih1.T + (b_ih1 + b_hh1)   (big WMMA GEMM, in-place over g0buf)
    {
        int groups = (BT / 64) * (GG / 16);   // 15360 (64x16 strips)
        wmma_gemm_bf16<<<groups / 4, 128, 0, stream>>>(
            y0buf, wih1bf, bias1, g0buf, BT, GG, HH, 0);
    }
    // 5. LSTM layer 1 (WMMA recurrence), writes hlast only
    lstm_rec_kernel<<<NB / 64, 256, lstm_smem, stream>>>(g0buf, W_hh1, nullptr, hlast);
    // 6. feature pipe -> feat[b]
    feat_kernel<<<NB, 64, 0, stream>>>(hlast, fp_w1, fp_b1, fp_w2, fp_b2, feat);
    // 7. neuron activations (bf16, padded)
    {
        size_t total = (size_t)NB * NPAD;
        neuron_kernel<<<(unsigned)((total + 255) / 256), 256, 0, stream>>>(
            feat, cs, sens, thr, neuron);
    }
    // 8. z1 = relu(neuron @ int_w1.T + int_b1)   (WMMA GEMM, K=2528)
    {
        int groups = (NB / 64) * (256 / 16);  // 128
        wmma_gemm_bf16<<<groups / 4, 128, 0, stream>>>(
            neuron, intw1bf, int_b1, z1, NB, 256, NPAD, 1);
    }
    // 9. tail: z2, integrated, heads, sigmoid(conf), group means
    head_kernel<<<NB, 256, 0, stream>>>(
        z1, int_w2, int_b2, int_w3, int_b3,
        tw, tb, pw, pb, kwp, kbp, vw, vb, cw, cb, neuron, out);
}
